// TrueHierarchicalHBNN_49022756716598
// MI455X (gfx1250) — compile-verified, matrix operand order checked by reading, not run on hardware
//
#include <hip/hip_runtime.h>
#include <stdint.h>

#define LOG2PI 1.8378770664093453f

typedef _Float16 half_t;
typedef __attribute__((ext_vector_type(16))) _Float16 v16h;
typedef __attribute__((ext_vector_type(8)))  float    v8f;
typedef int b128_t __attribute__((vector_size(16)));   // 4 x i32 payload type

#define AS1 __attribute__((address_space(1)))
#define AS3 __attribute__((address_space(3)))

#if __has_builtin(__builtin_amdgcn_global_load_async_to_lds_b128)
#define HAVE_ASYNC_LDS 1
#else
#define HAVE_ASYNC_LDS 0
#endif

// ---------- helpers ----------
__device__ __forceinline__ float sp(float x) {             // softplus
    return fmaxf(x, 0.0f) + log1pf(__expf(-fabsf(x)));
}

__device__ __forceinline__ uint32_t pcg(uint32_t v) {
    v = v * 747796405u + 2891336453u;
    uint32_t w = ((v >> ((v >> 28u) + 4u)) ^ v) * 277803737u;
    return (w >> 22u) ^ w;
}

// Deterministic standard-normal hash (Box-Muller). Not bit-identical to JAX
// threefry (impossible to match), but deterministic per call as required.
__device__ __forceinline__ float eps_normal(uint32_t layer, uint32_t b,
                                            uint32_t o, uint32_t i) {
    uint32_t s1 = pcg(layer * 0x9E3779B9u ^ pcg(b * 0x85EBCA6Bu ^ pcg(o * 0xC2B2AE35u ^ i)));
    uint32_t s2 = pcg(s1 ^ 0x68BC21EBu);
    float u1 = (s1 >> 8) * 0x1p-24f + 0x1p-25f;   // (0,1)
    float u2 = (s2 >> 8) * 0x1p-24f;
    float r  = __builtin_sqrtf(-2.0f * __logf(u1));
    return r * __cosf(6.28318530718f * u2);
}

__device__ __forceinline__ void wait_async0() {
#if HAVE_ASYNC_LDS
#if __has_builtin(__builtin_amdgcn_s_wait_asynccnt)
    __builtin_amdgcn_s_wait_asynccnt(0);
#else
    asm volatile("s_wait_asynccnt 0x0" ::: "memory");
#endif
#endif
}

// ---------- ARD scale: s[i] = softplus(a[i]) * softplus(b[i]) ----------
__global__ void ard_scale_kernel(const float* __restrict__ a,
                                 const float* __restrict__ b,
                                 float* __restrict__ s, int n) {
    int i = blockIdx.x * blockDim.x + threadIdx.x;
    if (i < n) s[i] = sp(a[i]) * sp(b[i]);
}

// ---------- fused hidden layer (WMMA term1 + VALU group/random terms) ----------
// Block = 128 threads (4 waves). blockIdx.x = 16-sample tile, blockIdx.y = 64-out tile.
// Each wave owns one 16x16 (samples x outs) tile.
template<int IN, int OUT, bool RELU>
__global__ void __launch_bounds__(128) layer_kernel(
    const float* __restrict__ src,  const int* __restrict__ gid,
    const float* __restrict__ gw_mu, const float* __restrict__ gw_rho,
    const float* __restrict__ gb_mu, const float* __restrict__ gb_rho,
    const float* __restrict__ rw_mu, const float* __restrict__ rw_rho,
    const float* __restrict__ rb_mu, const float* __restrict__ rb_rho,
    const float* __restrict__ scale, float* __restrict__ dst, uint32_t layer)
{
    constexpr int PAD = 4;                 // keep rows 16B-aligned, break bank stride
    constexpr int LDSTRIDE = IN + PAD;     // 2064B mod 256 = 16 -> distinct banks/row
    __shared__ float xs[16 * LDSTRIDE];
    __shared__ float ctile[4][16 * 16];
    __shared__ int   gtile[16];

    const int tid   = threadIdx.x;
    const int wave  = tid >> 5;
    const int lane  = tid & 31;
    const int btile = blockIdx.x;
    const int obase = blockIdx.y * 64 + wave * 16;

#if HAVE_ASYNC_LDS
    // Stage the 16xIN tile (contiguous in global) via async global->LDS B128,
    // tracked by ASYNCcnt; then apply the ARD scale in-LDS.
    {
        const float* gbase = src + (size_t)(btile * 16) * IN;
        constexpr int CHUNKS = 16 * IN / 4;         // float4 chunks
        for (int e = tid; e < CHUNKS; e += 128) {
            int flat = e * 4;
            int r = flat / IN, c = flat - r * IN;
            __builtin_amdgcn_global_load_async_to_lds_b128(
                (AS1 b128_t*)(gbase + flat),
                (AS3 b128_t*)&xs[r * LDSTRIDE + c], 0, 0);
        }
        wait_async0();
        if (tid < 16) gtile[tid] = gid[btile * 16 + tid];
        __syncthreads();
        for (int idx = tid; idx < 16 * IN; idx += 128) {
            int r = idx / IN, c = idx - r * IN;
            xs[r * LDSTRIDE + c] *= scale[c];
        }
    }
#else
    for (int idx = tid; idx < 16 * IN; idx += 128) {
        int r = idx / IN, c = idx - r * IN;
        xs[r * LDSTRIDE + c] = src[(btile * 16 + r) * IN + c] * scale[c];
    }
    if (tid < 16) gtile[tid] = gid[btile * 16 + tid];
#endif
    __syncthreads();

    // --- term1 via v_wmma_f32_16x16x32_f16: D = xs_tile(16xK) @ gw_mu^T(Kx16) ---
    v8f c = {};
    const int  Mrow = lane & 15;
    const bool hi   = lane >= 16;
    for (int k0 = 0; k0 < IN; k0 += 32) {
        v16h a, bfrag;
#pragma unroll
        for (int h = 0; h < 16; ++h) {
            int d = h >> 1, pos = h & 1;
            int kk = (d < 4 ? 2 * d + pos : 16 + 2 * (d - 4) + pos) + (hi ? 8 : 0);
            a[h]     = (half_t)xs[Mrow * LDSTRIDE + k0 + kk];
            bfrag[h] = (half_t)gw_mu[(size_t)(obase + Mrow) * IN + k0 + kk];
        }
        c = __builtin_amdgcn_wmma_f32_16x16x32_f16(false, a, false, bfrag,
                                                   (short)0, c, false, false);
    }
#pragma unroll
    for (int r = 0; r < 8; ++r)   // C layout: lanes<16 -> M=r, lanes>=16 -> M=r+8
        ctile[wave][(r + (hi ? 8 : 0)) * 16 + (lane & 15)] = c[r];
    __syncthreads();

    // --- VALU: group + random terms, streaming rw_mu/rw_rho (memory-bound) ---
    for (int it = 0; it < 8; ++it) {
        int p  = it * 32 + lane;
        int bl = p & 15;
        int ol = p >> 4;
        int b  = btile * 16 + bl;
        int o  = obase + ol;
        int g  = gtile[bl];
        const float4* rmu4  = reinterpret_cast<const float4*>(rw_mu  + ((size_t)g * OUT + o) * IN);
        const float4* rrho4 = reinterpret_cast<const float4*>(rw_rho + ((size_t)g * OUT + o) * IN);
        const float4* grho4 = reinterpret_cast<const float4*>(gw_rho + (size_t)o * IN);
        float acc = ctile[wave][bl * 16 + ol];
        for (int i4 = 0; i4 < IN / 4; ++i4) {
            if ((i4 & 15) == 0) {
                __builtin_prefetch(rmu4  + i4 + 16, 0, 1);   // global_prefetch_b8
                __builtin_prefetch(rrho4 + i4 + 16, 0, 1);
            }
            float4 rmv = rmu4[i4];
            float4 rrv = rrho4[i4];
            float4 grv = grho4[i4];
            float rm[4] = {rmv.x, rmv.y, rmv.z, rmv.w};
            float rr[4] = {rrv.x, rrv.y, rrv.z, rrv.w};
            float gr[4] = {grv.x, grv.y, grv.z, grv.w};
#pragma unroll
            for (int u = 0; u < 4; ++u) {
                int i = i4 * 4 + u;
                float xv  = xs[bl * LDSTRIDE + i];
                float gws = sp(gr[u]);
                float rws = sp(rr[u]);
                float e   = eps_normal(layer, (uint32_t)b, (uint32_t)o, (uint32_t)i);
                acc += xv * gws * fmaf(rws, e, rm[u]);
            }
        }
        // bias
        float gbs = sp(gb_rho[o]);
        float rbs = sp(rb_rho[(size_t)g * OUT + o]);
        float eb  = eps_normal(layer ^ 0xB1A50000u, (uint32_t)b, (uint32_t)o, 0u);
        acc += gb_mu[o] + gbs * fmaf(rbs, eb, rb_mu[(size_t)g * OUT + o]);
        if (RELU) acc = fmaxf(acc, 0.0f);
        dst[(size_t)b * OUT + o] = acc;
    }
}

// ---------- output layer (OUT = 1): plain per-sample dot ----------
__global__ void __launch_bounds__(256) layer_out_kernel(
    const float* __restrict__ src,  const int* __restrict__ gid,
    const float* __restrict__ gw_mu, const float* __restrict__ gw_rho,
    const float* __restrict__ gb_mu, const float* __restrict__ gb_rho,
    const float* __restrict__ rw_mu, const float* __restrict__ rw_rho,
    const float* __restrict__ rb_mu, const float* __restrict__ rb_rho,
    const float* __restrict__ scale, float* __restrict__ dst, uint32_t layer)
{
    int b = blockIdx.x * 256 + threadIdx.x;
    if (b >= 512) return;
    int g = gid[b];
    const float4* rmu4  = reinterpret_cast<const float4*>(rw_mu  + (size_t)g * 512);
    const float4* rrho4 = reinterpret_cast<const float4*>(rw_rho + (size_t)g * 512);
    const float4* gmu4  = reinterpret_cast<const float4*>(gw_mu);
    const float4* grho4 = reinterpret_cast<const float4*>(gw_rho);
    const float4* src4  = reinterpret_cast<const float4*>(src + (size_t)b * 512);
    const float4* sc4   = reinterpret_cast<const float4*>(scale);
    float acc = 0.0f;
    for (int i4 = 0; i4 < 128; ++i4) {
        float4 rmv = rmu4[i4], rrv = rrho4[i4], gmv = gmu4[i4], grv = grho4[i4];
        float4 xvv = src4[i4], scv = sc4[i4];
        float rm[4] = {rmv.x, rmv.y, rmv.z, rmv.w};
        float rr[4] = {rrv.x, rrv.y, rrv.z, rrv.w};
        float gm[4] = {gmv.x, gmv.y, gmv.z, gmv.w};
        float gr[4] = {grv.x, grv.y, grv.z, grv.w};
        float xv[4] = {xvv.x, xvv.y, xvv.z, xvv.w};
        float sc[4] = {scv.x, scv.y, scv.z, scv.w};
#pragma unroll
        for (int u = 0; u < 4; ++u) {
            int i = i4 * 4 + u;
            float x  = xv[u] * sc[u];
            float gws = sp(gr[u]);
            float rws = sp(rr[u]);
            float e   = eps_normal(layer, (uint32_t)b, 0u, (uint32_t)i);
            acc += x * (gm[u] + gws * fmaf(rws, e, rm[u]));
        }
    }
    float gbs = sp(gb_rho[0]);
    float rbs = sp(rb_rho[g]);
    float eb  = eps_normal(layer ^ 0xB1A50000u, (uint32_t)b, 0u, 0u);
    acc += gb_mu[0] + gbs * fmaf(rbs, eb, rb_mu[g]);
    dst[b] = acc;
}

// ---------- KL reduction ----------
// mode 0: group_kl + hyp(gws)      over (gw_mu, gw_rho)
// mode 1: ind_kl                   over (rw_mu, rw_rho)
// mode 2: hyp                      over gb_rho
// mode 3: sum(softplus)            over ard_a / ard_b
__global__ void kl_kernel(const float* __restrict__ mu,
                          const float* __restrict__ rho,
                          int n, int mode, float* __restrict__ out) {
    float local = 0.0f;
    for (int idx = blockIdx.x * blockDim.x + threadIdx.x; idx < n;
         idx += gridDim.x * blockDim.x) {
        float acc;
        if (mode == 3) {
            acc = sp(rho[idx]);
        } else {
            float t  = sp(rho[idx]);
            float lt = __logf(t);
            if (mode == 0) {
                float m = mu[idx];
                acc = (-lt + 0.5f * (t * t + m * m) - 0.5f)
                    + (-0.5f * LOG2PI - lt + 0.5f * (lt * lt + t * t) - 0.5f);
            } else if (mode == 1) {
                float m = mu[idx];
                acc = -0.5f * LOG2PI - lt + 0.5f * (t * t + m * m) - 0.5f;
            } else {
                acc = -0.5f * LOG2PI - lt + 0.5f * (lt * lt + t * t) - 0.5f;
            }
        }
        local += acc;
    }
    // wave32 reduction
    for (int off = 16; off > 0; off >>= 1)
        local += __shfl_down(local, off, 32);
    __shared__ float wsum[8];
    int lane = threadIdx.x & 31, wv = threadIdx.x >> 5;
    if (lane == 0) wsum[wv] = local;
    __syncthreads();
    if (threadIdx.x == 0) {
        float s = 0.0f;
        int nw = (blockDim.x + 31) / 32;
        for (int w = 0; w < nw; ++w) s += wsum[w];
        atomicAdd(out, s);   // global_atomic_add_f32
    }
}

__global__ void zero_one_kernel(float* p) { *p = 0.0f; }

// ---------- launch ----------
extern "C" void kernel_launch(void* const* d_in, const int* in_sizes, int n_in,
                              void* d_out, int out_size, void* d_ws, size_t ws_size,
                              hipStream_t stream) {
    const float* x   = (const float*)d_in[0];
    const int*   gid = (const int*)d_in[1];
    const float* P[3][10];
    for (int l = 0; l < 3; ++l)
        for (int k = 0; k < 10; ++k)
            P[l][k] = (const float*)d_in[2 + 10 * l + k];
    // param order per layer: gw_mu, gw_rho, gb_mu, gb_rho, rw_mu, rw_rho, rb_mu, rb_rho, ard_a, ard_b

    float* out = (float*)d_out;           // [0..511] = h, [512] = kl
    float* W   = (float*)d_ws;
    float* h0 = W;                        // 512*512
    float* h1 = W + 262144;               // 512*512
    float* s0 = W + 524288;               // 256
    float* s1 = s0 + 256;                 // 512
    float* s2 = s1 + 512;                 // 512

    // ARD scales
    ard_scale_kernel<<<1, 256, 0, stream>>>(P[0][8], P[0][9], s0, 256);
    ard_scale_kernel<<<2, 256, 0, stream>>>(P[1][8], P[1][9], s1, 512);
    ard_scale_kernel<<<2, 256, 0, stream>>>(P[2][8], P[2][9], s2, 512);

    // layer 0: 256 -> 512, relu
    layer_kernel<256, 512, true><<<dim3(32, 8), 128, 0, stream>>>(
        x, gid, P[0][0], P[0][1], P[0][2], P[0][3], P[0][4], P[0][5],
        P[0][6], P[0][7], s0, h0, 0u);
    // layer 1: 512 -> 512, relu
    layer_kernel<512, 512, true><<<dim3(32, 8), 128, 0, stream>>>(
        h0, gid, P[1][0], P[1][1], P[1][2], P[1][3], P[1][4], P[1][5],
        P[1][6], P[1][7], s1, h1, 1u);
    // layer 2: 512 -> 1
    layer_out_kernel<<<2, 256, 0, stream>>>(
        h1, gid, P[2][0], P[2][1], P[2][2], P[2][3], P[2][4], P[2][5],
        P[2][6], P[2][7], s2, out, 2u);

    // KL
    zero_one_kernel<<<1, 1, 0, stream>>>(out + 512);
    const int dims_in[3]  = {256, 512, 512};
    const int dims_out[3] = {512, 512, 1};
    for (int l = 0; l < 3; ++l) {
        int ngw = dims_out[l] * dims_in[l];
        int nrw = 64 * ngw;
        int bg  = (ngw + 255) / 256; if (bg > 512) bg = 512;
        int br  = (nrw + 255) / 256; if (br > 512) br = 512;
        kl_kernel<<<bg, 256, 0, stream>>>(P[l][0], P[l][1], ngw, 0, out + 512);
        kl_kernel<<<br, 256, 0, stream>>>(P[l][4], P[l][5], nrw, 1, out + 512);
        kl_kernel<<<1, 256, 0, stream>>>(P[l][3], P[l][3], dims_out[l], 2, out + 512);
        kl_kernel<<<1, 256, 0, stream>>>(P[l][8], P[l][8], dims_in[l], 3, out + 512);
        kl_kernel<<<1, 256, 0, stream>>>(P[l][9], P[l][9], dims_in[l], 3, out + 512);
    }
}
